// SubsetMaxMatchesTTADualSoftMaxMatcher_61014305407067
// MI455X (gfx1250) — matrix operand order repaired and checked by reading, not compiled, not run on hardware
//
#include <hip/hip_runtime.h>
#include <hip/hip_bf16.h>

// ---------------------------------------------------------------------------
// CDNA5 (gfx1250) dual-softmax mutual matcher.
// All heavy math runs through v_wmma_f32_16x16x32_f16 (wave32 WMMA).
// ---------------------------------------------------------------------------

typedef _Float16 v16h __attribute__((ext_vector_type(16)));
typedef _Float16 v8h  __attribute__((ext_vector_type(8)));
typedef float    v8f  __attribute__((ext_vector_type(8)));

#define INV_TEMP 20.0f
#define THRESH   0.01f
#define K_DIM    256
#define N_FULL   12288
#define N_SUB    4096
#define N_AFF    8

// Monotone float -> uint key (order-preserving), so atomicMax works exactly.
__device__ __forceinline__ unsigned fkey(float f) {
  unsigned u = __float_as_uint(f);
  return (u & 0x80000000u) ? ~u : (u | 0x80000000u);
}
__device__ __forceinline__ float funkey(unsigned k) {
  return (k & 0x80000000u) ? __uint_as_float(k ^ 0x80000000u)
                           : __uint_as_float(~k);
}
// Must be bitwise-identical between pass 3 and pass 4.
__device__ __forceinline__ float computeP(float s, float rm, float rs,
                                          float cm, float cs) {
  float er = __expf(s - rm);
  float ec = __expf(s - cm);
  return (er / rs) * (ec / cs);
}

// ---------------------------------------------------------------------------
// f32 -> f16 conversion of B descriptors (full + strided subset)
// ---------------------------------------------------------------------------
__global__ __launch_bounds__(256)
void k_convertB(const float* __restrict__ dB, _Float16* __restrict__ dB16,
                _Float16* __restrict__ dBsub16) {
  int idx = blockIdx.x * 256 + threadIdx.x;       // [0, 12288*256)
  float v = dB[idx];
  _Float16 h = (_Float16)v;
  dB16[idx] = h;
  int row = idx >> 8, k = idx & 255;
  if (row % 3 == 0) dBsub16[((row / 3) << 8) | k] = h;
}

// ---------------------------------------------------------------------------
// Steer subset rows (stride 3) for all 8 affine candidates, emit f16
// ---------------------------------------------------------------------------
__global__ __launch_bounds__(256)
void k_steerSub(const float* __restrict__ dA, const float* __restrict__ affs,
                _Float16* __restrict__ out) {
  int tid = blockIdx.x * 256 + threadIdx.x;       // [0, 8*4096*128)
  int a   = tid >> 19;
  int rem = tid & ((1 << 19) - 1);
  int n   = rem >> 7;
  int kp  = rem & 127;
  const float* aff = affs + a * 4;
  size_t src = (size_t)(n * 3) * K_DIM + kp * 2;
  float d0 = dA[src], d1 = dA[src + 1];
  size_t dst = ((size_t)a * N_SUB + n) * K_DIM + kp * 2;
  out[dst]     = (_Float16)(aff[0] * d0 + aff[1] * d1);
  out[dst + 1] = (_Float16)(aff[2] * d0 + aff[3] * d1);
}

// Steer full A with the on-device best affine
__global__ __launch_bounds__(256)
void k_steerFull(const float* __restrict__ dA, const float* __restrict__ affs,
                 const int* __restrict__ best, _Float16* __restrict__ out) {
  int tid = blockIdx.x * 256 + threadIdx.x;       // [0, 12288*128)
  int n  = tid >> 7;
  int kp = tid & 127;
  const float* aff = affs + (*best) * 4;
  size_t src = (size_t)n * K_DIM + kp * 2;
  float d0 = dA[src], d1 = dA[src + 1];
  out[src]     = (_Float16)(aff[0] * d0 + aff[1] * d1);
  out[src + 1] = (_Float16)(aff[2] * d0 + aff[3] * d1);
}

// ---------------------------------------------------------------------------
// Core WMMA GEMM + fused dual-softmax statistics.
//  PASS 1: row/col max of S            PASS 2: row/col sum of exp(S-max) (partials)
//  PASS 3: row/col max of P            PASS 4: mutual-match test / argmax / count
// Block = 256 threads = 8 wave32 waves; block tile 128x128; wave tile 16x128.
// ---------------------------------------------------------------------------
template <int PASS, bool SUBSET>
__global__ __launch_bounds__(256)
void k_gemm(const _Float16* __restrict__ A, const _Float16* __restrict__ B,
            int N, int rowsPerProblem,
            unsigned* __restrict__ rowMaxKey, float* __restrict__ rowSum,
            unsigned* __restrict__ rowPMaxKey,
            unsigned* __restrict__ colMaxKey, float* __restrict__ colSum,
            unsigned* __restrict__ colPMaxKey,
            float* __restrict__ rowSumPart, float* __restrict__ colSumPart,
            int* __restrict__ counts, int* __restrict__ colIdx,
            int* __restrict__ rowMask) {
  __shared__ unsigned scolU[128];
  __shared__ float    scolW[8 * 128];
  __shared__ int      scnt;

  const int lane  = threadIdx.x & 31;
  const int wv    = threadIdx.x >> 5;
  const int laneM = lane & 15;
  const int half  = lane >> 4;                    // 0 or 1
  const int blockRowBase = blockIdx.y * 128;
  const int blockColBase = blockIdx.x * 128;

  v8f acc[8];
#pragma unroll
  for (int nt = 0; nt < 8; ++nt)
#pragma unroll
    for (int e = 0; e < 8; ++e) acc[nt][e] = 0.0f;

  // A-frag (16x32 f16): lanes 0-15 hold K 0..15 contiguous, lanes 16-31 K 16..31
  // B-frag (32x16 f16): lanes 0-15 hold K 0..15 contiguous, lanes 16-31 K 16..31
  const size_t arowOff = (size_t)(blockRowBase + wv * 16 + laneM) * K_DIM;
  const size_t browOff = (size_t)(blockColBase + laneM) * K_DIM;
  const int ka = half * 16;
  const int kb = half * 16;

#pragma unroll
  for (int kk = 0; kk < K_DIM / 32; ++kk) {
    const _Float16* ap = A + arowOff + kk * 32 + ka;
    v8h lo = *reinterpret_cast<const v8h*>(ap);
    v8h hi = *reinterpret_cast<const v8h*>(ap + 8);
    v16h af;
#pragma unroll
    for (int j = 0; j < 8; ++j) { af[j] = lo[j]; af[j + 8] = hi[j]; }

    v16h bf[8];
#pragma unroll
    for (int nt = 0; nt < 8; ++nt)
      bf[nt] = *reinterpret_cast<const v16h*>(
          B + browOff + (size_t)nt * (16 * K_DIM) + kk * 32 + kb);
#pragma unroll
    for (int nt = 0; nt < 8; ++nt)
      acc[nt] = __builtin_amdgcn_wmma_f32_16x16x32_f16(
          false, af, false, bf[nt], (short)0, acc[nt], false, false);
  }

  // C/D layout: element e, lanes 0-15 -> (M=e, N=lane); lanes 16-31 -> (M=8+e)
  const int rowG0  = blockRowBase + wv * 16 + half * 8;
  const int aff    = blockRowBase / rowsPerProblem;
  const int affOff = aff * N;

  if constexpr (PASS == 1 || PASS == 3) {
    if constexpr (PASS == 1) {
#pragma unroll
      for (int nt = 0; nt < 8; ++nt)
#pragma unroll
        for (int e = 0; e < 8; ++e) acc[nt][e] = INV_TEMP * acc[nt][e];
    } else {
      float rmaxF[8], rsumF[8];
#pragma unroll
      for (int e = 0; e < 8; ++e) {
        rmaxF[e] = funkey(rowMaxKey[rowG0 + e]);
        rsumF[e] = rowSum[rowG0 + e];
      }
      float cmaxF[8], csumF[8];
#pragma unroll
      for (int nt = 0; nt < 8; ++nt) {
        int cg = affOff + blockColBase + nt * 16 + laneM;
        cmaxF[nt] = funkey(colMaxKey[cg]);
        csumF[nt] = colSum[cg];
      }
#pragma unroll
      for (int nt = 0; nt < 8; ++nt)
#pragma unroll
        for (int e = 0; e < 8; ++e)
          acc[nt][e] = computeP(INV_TEMP * acc[nt][e], rmaxF[e], rsumF[e],
                                cmaxF[nt], csumF[nt]);
    }
    unsigned* rDst = (PASS == 1) ? rowMaxKey : rowPMaxKey;
    unsigned* cDst = (PASS == 1) ? colMaxKey : colPMaxKey;

    float rmx[8];
#pragma unroll
    for (int e = 0; e < 8; ++e) {
      float m = -3.402823466e38f;
#pragma unroll
      for (int nt = 0; nt < 8; ++nt) m = fmaxf(m, acc[nt][e]);
      rmx[e] = m;
    }
#pragma unroll
    for (int off = 8; off >= 1; off >>= 1)
#pragma unroll
      for (int e = 0; e < 8; ++e)
        rmx[e] = fmaxf(rmx[e], __shfl_xor(rmx[e], off, 32));
    if (laneM == 0) {
#pragma unroll
      for (int e = 0; e < 8; ++e) atomicMax(&rDst[rowG0 + e], fkey(rmx[e]));
    }
    if (threadIdx.x < 128) scolU[threadIdx.x] = 0u;
    __syncthreads();
#pragma unroll
    for (int nt = 0; nt < 8; ++nt) {
      float cm = -3.402823466e38f;
#pragma unroll
      for (int e = 0; e < 8; ++e) cm = fmaxf(cm, acc[nt][e]);
      cm = fmaxf(cm, __shfl_xor(cm, 16, 32));
      if (lane < 16) atomicMax(&scolU[nt * 16 + lane], fkey(cm));
    }
    __syncthreads();
    if (threadIdx.x < 128)
      atomicMax(&cDst[affOff + blockColBase + threadIdx.x], scolU[threadIdx.x]);
  } else if constexpr (PASS == 2) {
    float rmaxF[8], rs[8];
#pragma unroll
    for (int e = 0; e < 8; ++e) {
      rmaxF[e] = funkey(rowMaxKey[rowG0 + e]);
      rs[e] = 0.f;
    }
    float cmaxF[8], cs[8];
#pragma unroll
    for (int nt = 0; nt < 8; ++nt) {
      cmaxF[nt] = funkey(colMaxKey[affOff + blockColBase + nt * 16 + laneM]);
      cs[nt] = 0.f;
    }
#pragma unroll
    for (int nt = 0; nt < 8; ++nt)
#pragma unroll
      for (int e = 0; e < 8; ++e) {
        float s = INV_TEMP * acc[nt][e];
        rs[e]  += __expf(s - rmaxF[e]);
        cs[nt] += __expf(s - cmaxF[nt]);
      }
#pragma unroll
    for (int off = 8; off >= 1; off >>= 1)
#pragma unroll
      for (int e = 0; e < 8; ++e) rs[e] += __shfl_xor(rs[e], off, 32);
    if (laneM == 0) {
#pragma unroll
      for (int e = 0; e < 8; ++e)
        rowSumPart[(size_t)(rowG0 + e) * gridDim.x + blockIdx.x] = rs[e];
    }
#pragma unroll
    for (int nt = 0; nt < 8; ++nt) {
      float c = cs[nt] + __shfl_xor(cs[nt], 16, 32);
      if (lane < 16) scolW[wv * 128 + nt * 16 + lane] = c;
    }
    __syncthreads();
    if (threadIdx.x < 128) {
      float tot = 0.f;
#pragma unroll
      for (int w2 = 0; w2 < 8; ++w2) tot += scolW[w2 * 128 + threadIdx.x];
      int rbl = blockIdx.y % gridDim.x;  // row-block index within this problem
      colSumPart[(size_t)(affOff + blockColBase + threadIdx.x) * gridDim.x +
                 rbl] = tot;
    }
  } else {  // PASS == 4
    float rmaxF[8], rsumF[8];
    unsigned rPK[8];
#pragma unroll
    for (int e = 0; e < 8; ++e) {
      rmaxF[e] = funkey(rowMaxKey[rowG0 + e]);
      rsumF[e] = rowSum[rowG0 + e];
      rPK[e]   = rowPMaxKey[rowG0 + e];
    }
    float cmaxF[8], csumF[8];
    unsigned cPK[8];
#pragma unroll
    for (int nt = 0; nt < 8; ++nt) {
      int cg = affOff + blockColBase + nt * 16 + laneM;
      cmaxF[nt] = funkey(colMaxKey[cg]);
      csumF[nt] = colSum[cg];
      cPK[nt]   = colPMaxKey[cg];
    }
    int cnt = 0;
#pragma unroll
    for (int nt = 0; nt < 8; ++nt)
#pragma unroll
      for (int e = 0; e < 8; ++e) {
        float p = computeP(INV_TEMP * acc[nt][e], rmaxF[e], rsumF[e],
                           cmaxF[nt], csumF[nt]);
        unsigned kp = fkey(p);
        bool rmHit = (kp == rPK[e]);
        bool cmHit = (kp == cPK[nt]);
        bool mut   = rmHit && cmHit && (p > THRESH);
        if constexpr (SUBSET) {
          cnt += mut ? 1 : 0;
        } else {
          int rg  = rowG0 + e;
          int cg2 = blockColBase + nt * 16 + laneM;
          if (rmHit) atomicMin(&colIdx[rg], cg2);
          if (mut) rowMask[rg] = 1;
        }
      }
    if constexpr (SUBSET) {
      if (threadIdx.x == 0) scnt = 0;
      __syncthreads();
      atomicAdd(&scnt, cnt);
      __syncthreads();
      if (threadIdx.x == 0) atomicAdd(&counts[aff], scnt);
    }
  }
}

// Fixed-order reduction of row/col softmax-sum partials (deterministic)
__global__ __launch_bounds__(256)
void k_reduce(const float* __restrict__ rowSumPart,
              const float* __restrict__ colSumPart, float* __restrict__ rowSum,
              float* __restrict__ colSum, int Mtotal, int nb) {
  int i = blockIdx.x * 256 + threadIdx.x;
  if (i >= Mtotal) return;
  float r = 0.f, c = 0.f;
  for (int b = 0; b < nb; ++b) {
    r += rowSumPart[(size_t)i * nb + b];
    c += colSumPart[(size_t)i * nb + b];
  }
  rowSum[i] = r;
  colSum[i] = c;
}

__global__ void k_best(const int* __restrict__ counts, int* __restrict__ best,
                       float* __restrict__ outBest) {
  int bi = 0, bc = counts[0];
  for (int a = 1; a < N_AFF; ++a) {
    int c = counts[a];
    if (c > bc) { bc = c; bi = a; }
  }
  *best = bi;
  *outBest = (float)bi;
}

__global__ __launch_bounds__(256)
void k_final(const float* __restrict__ kpA, const float* __restrict__ kpB,
             const int* __restrict__ colIdx, const int* __restrict__ rowMask,
             float* __restrict__ out) {
  int i = blockIdx.x * 256 + threadIdx.x;
  if (i >= N_FULL) return;
  float m = rowMask[i] ? 1.f : 0.f;
  int ci = colIdx[i];
  if ((unsigned)ci >= (unsigned)N_FULL) ci = 0;
  out[2 * i]     = kpA[2 * i] * m;
  out[2 * i + 1] = kpA[2 * i + 1] * m;
  float* outB = out + 2 * N_FULL;
  outB[2 * i]     = kpB[2 * ci] * m;
  outB[2 * i + 1] = kpB[2 * ci + 1] * m;
  out[4 * N_FULL + i] = m;
  out[5 * N_FULL + i] = (float)ci;
}

// ---------------------------------------------------------------------------
extern "C" void kernel_launch(void* const* d_in, const int* in_sizes, int n_in,
                              void* d_out, int out_size, void* d_ws,
                              size_t ws_size, hipStream_t stream) {
  (void)in_sizes; (void)n_in; (void)ws_size;
  const float* kpA  = (const float*)d_in[0];
  const float* dA   = (const float*)d_in[1];
  const float* kpB  = (const float*)d_in[2];
  const float* dB   = (const float*)d_in[3];
  const float* affs = (const float*)d_in[4];
  float* out = (float*)d_out;

  // ---- workspace carve-up (all 256B aligned) ----
  char* w = (char*)d_ws;
  auto carve = [&](size_t bytes) {
    char* p = w;
    w += (bytes + 255) & ~(size_t)255;
    return p;
  };
  _Float16* dB16    = (_Float16*)carve((size_t)N_FULL * K_DIM * 2);
  _Float16* dBsub16 = (_Float16*)carve((size_t)N_SUB * K_DIM * 2);
  _Float16* dAsubSt = (_Float16*)carve((size_t)N_AFF * N_SUB * K_DIM * 2);
  _Float16* dAfull  = (_Float16*)carve((size_t)N_FULL * K_DIM * 2);
  const int STAT = N_AFF * N_SUB;  // 32768 >= 12288, reused for full phase
  unsigned* rowMaxKey  = (unsigned*)carve((size_t)STAT * 4);
  float*    rowSum     = (float*)carve((size_t)STAT * 4);
  unsigned* rowPMaxKey = (unsigned*)carve((size_t)STAT * 4);
  unsigned* colMaxKey  = (unsigned*)carve((size_t)STAT * 4);
  float*    colSum     = (float*)carve((size_t)STAT * 4);
  unsigned* colPMaxKey = (unsigned*)carve((size_t)STAT * 4);
  int* counts  = (int*)carve(64);
  int* bestIdx = counts + 8;  // lives inside the same 64B slot
  int* colIdx  = (int*)carve((size_t)N_FULL * 4);
  int* rowMask = (int*)carve((size_t)N_FULL * 4);
  const size_t PARTS = (size_t)N_FULL * (N_FULL / 128);  // 12288*96 (max need)
  float* rowSumPart = (float*)carve(PARTS * 4);
  float* colSumPart = (float*)carve(PARTS * 4);
  const size_t statsBytes = 6 * (size_t)STAT * 4;  // the 6 arrays are contiguous

  // ---- prep: convert B, steer subset for all 8 affs ----
  k_convertB<<<N_FULL * K_DIM / 256, 256, 0, stream>>>(dB, dB16, dBsub16);
  k_steerSub<<<(N_AFF * N_SUB * 128) / 256, 256, 0, stream>>>(dA, affs, dAsubSt);
  hipMemsetAsync(rowMaxKey, 0, statsBytes, stream);
  hipMemsetAsync(counts, 0, 64, stream);

  // ---- subset phase: batched M = 8*4096, N = 4096, K = 256 ----
  dim3 gs(N_SUB / 128, (N_AFF * N_SUB) / 128);
#define GEMM_ARGS_SUB dAsubSt, dBsub16, N_SUB, N_SUB, rowMaxKey, rowSum,      \
    rowPMaxKey, colMaxKey, colSum, colPMaxKey, rowSumPart, colSumPart,        \
    counts, colIdx, rowMask
  k_gemm<1, true><<<gs, 256, 0, stream>>>(GEMM_ARGS_SUB);
  k_gemm<2, true><<<gs, 256, 0, stream>>>(GEMM_ARGS_SUB);
  k_reduce<<<(STAT + 255) / 256, 256, 0, stream>>>(rowSumPart, colSumPart,
                                                   rowSum, colSum, STAT,
                                                   N_SUB / 128);
  k_gemm<3, true><<<gs, 256, 0, stream>>>(GEMM_ARGS_SUB);
  k_gemm<4, true><<<gs, 256, 0, stream>>>(GEMM_ARGS_SUB);

  // ---- pick best affine on-device, steer full A ----
  k_best<<<1, 1, 0, stream>>>(counts, bestIdx, out + (out_size - 1));
  k_steerFull<<<(N_FULL * 128) / 256, 256, 0, stream>>>(dA, affs, bestIdx,
                                                        dAfull);
  hipMemsetAsync(rowMaxKey, 0, statsBytes, stream);
  hipMemsetAsync(colIdx, 0x7F, (size_t)N_FULL * 4, stream);  // large sentinel
  hipMemsetAsync(rowMask, 0, (size_t)N_FULL * 4, stream);

  // ---- full phase: M = N = 12288, K = 256 ----
  dim3 gf(N_FULL / 128, N_FULL / 128);
#define GEMM_ARGS_FULL dAfull, dB16, N_FULL, N_FULL, rowMaxKey, rowSum,       \
    rowPMaxKey, colMaxKey, colSum, colPMaxKey, rowSumPart, colSumPart,        \
    counts, colIdx, rowMask
  k_gemm<1, false><<<gf, 256, 0, stream>>>(GEMM_ARGS_FULL);
  k_gemm<2, false><<<gf, 256, 0, stream>>>(GEMM_ARGS_FULL);
  k_reduce<<<(N_FULL + 255) / 256, 256, 0, stream>>>(rowSumPart, colSumPart,
                                                     rowSum, colSum, N_FULL,
                                                     N_FULL / 128);
  k_gemm<3, false><<<gf, 256, 0, stream>>>(GEMM_ARGS_FULL);
  k_gemm<4, false><<<gf, 256, 0, stream>>>(GEMM_ARGS_FULL);

  // ---- emit outputs ----
  k_final<<<N_FULL / 256, 256, 0, stream>>>(kpA, kpB, colIdx, rowMask, out);
}
